// LoraInjectedLinear_75746043232770
// MI455X (gfx1250) — matrix operand order repaired and checked by reading, not compiled, not run on hardware
//
#include <hip/hip_runtime.h>

#define IN_F 1280
#define OUT_F 1280
#define RANK 4
#define LORA_STRIDE 4
#define TOKENS 4096
#define SCALE 1.0f

typedef __attribute__((ext_vector_type(16))) __bf16 v16bf;
typedef __attribute__((ext_vector_type(8)))  float  v8f;

union Frag { v16bf v; uint4 q[2]; };

// ---- packed fp32 -> bf16 helpers -------------------------------------------
// RNE scalar fallback
__device__ __forceinline__ unsigned rne16(unsigned u) {
    return (u + 0x7FFFu + ((u >> 16) & 1u)) >> 16;
}

// Pack two floats to a bf16 pair (low = f0, high = f1), RNE if HW packed cvt exists.
__device__ __forceinline__ unsigned pack_rne(float f0, float f1) {
#if __has_builtin(__builtin_amdgcn_cvt_pk_bf16_f32)
    auto p = __builtin_amdgcn_cvt_pk_bf16_f32(f0, f1);
    unsigned u; __builtin_memcpy(&u, &p, sizeof(u));
    return u;
#else
    union { float f; unsigned u; } a, b; a.f = f0; b.f = f1;
    return rne16(a.u) | (rne16(b.u) << 16);
#endif
}

// Truncating hi-pack (cheap; residual captured exactly by the lo term).
__device__ __forceinline__ unsigned pack_trunc(float f0, float f1) {
    union { float f; unsigned u; } a, b; a.f = f0; b.f = f1;
    return (a.u >> 16) | (b.u & 0xFFFF0000u);  // single v_perm_b32
}

// Split (f0,f1) into packed hi and lo bf16 pairs and store as dwords.
__device__ __forceinline__ void store_split(unsigned short* hp, unsigned short* lp,
                                            float f0, float f1) {
    unsigned hu = pack_trunc(f0, f1);
    union { unsigned u; float f; } h0, h1;
    h0.u = hu << 16;            // hi(f0) as fp32 bits
    h1.u = hu & 0xFFFF0000u;    // hi(f1) as fp32 bits
    unsigned lu = pack_rne(f0 - h0.f, f1 - h1.f);
    *(unsigned*)hp = hu;
    *(unsigned*)lp = lu;
}
// ----------------------------------------------------------------------------

// 128x128 output tile per workgroup; split-bf16 (hi/lo) WMMA emulating fp32 GEMM,
// with rank-4 LoRA (down+up) fused in.
__global__ __launch_bounds__(256)
void lora_gemm_kernel(const float* __restrict__ x,
                      const int*   __restrict__ lora_id,
                      const float* __restrict__ W,
                      const float* __restrict__ Wd,
                      const float* __restrict__ Wu,
                      float*       __restrict__ out)
{
    constexpr int S = 40;   // LDS row stride in bf16 elems (32 + 8 pad -> 80B, bank-conflict-free)

    __shared__ __align__(16) unsigned short sAh[128 * S];
    __shared__ __align__(16) unsigned short sAl[128 * S];
    __shared__ __align__(16) unsigned short sBh[128 * S];
    __shared__ __align__(16) unsigned short sBl[128 * S];
    __shared__ __align__(16) unsigned short sDh[16 * S];
    __shared__ __align__(16) unsigned short sDl[16 * S];
    __shared__ __align__(16) float sDown[128 * RANK];

    const int t    = threadIdx.x;
    const int lane = t & 31;
    const int l16  = lane & 15;
    const int lh   = lane >> 4;
    const int wave = t >> 5;
    const int wr   = wave >> 2;   // 0..1 : 64-row slab
    const int wc   = wave & 3;    // 0..3 : 32-col slab

    const int b    = blockIdx.z;
    const int row0 = blockIdx.y * 128;
    const int col0 = blockIdx.x * 128;

    const int  lid    = lora_id[b];
    const bool active = (lid >= 0);
    const int  safe   = active ? (lid / LORA_STRIDE) : 0;

    const float* xA = x  + ((size_t)b * TOKENS + row0) * IN_F;
    const float* wB = W  + (size_t)col0 * IN_F;
    const float* wD = Wd + (size_t)safe * RANK * IN_F;

    v8f acc[4][2];
    v8f dacc;
    #pragma unroll
    for (int i = 0; i < 4; ++i)
        #pragma unroll
        for (int j = 0; j < 2; ++j)
            #pragma unroll
            for (int e = 0; e < 8; ++e) acc[i][j][e] = 0.0f;
    #pragma unroll
    for (int e = 0; e < 8; ++e) dacc[e] = 0.0f;

    // cooperative-load geometry: 256 threads cover 128 rows x 32 cols fp32 in 4 passes
    const int lrow = t >> 3;          // 0..31
    const int lcol = (t & 7) * 4;     // 0,4,...,28
    const int drow = (t * 2) >> 5;    // 0..15 (Wd tile row, zero-padded past RANK)
    const int dcol = (t * 2) & 31;

    float pa[16], pb[16];
    float pd0 = 0.f, pd1 = 0.f;

    // prefetch K-slab 0
    #pragma unroll
    for (int p = 0; p < 4; ++p) {
        *(float4*)&pa[p * 4] = *(const float4*)&xA[(size_t)(lrow + 32 * p) * IN_F + lcol];
        *(float4*)&pb[p * 4] = *(const float4*)&wB[(size_t)(lrow + 32 * p) * IN_F + lcol];
    }
    if (drow < RANK) {
        float2 d = *(const float2*)&wD[(size_t)drow * IN_F + dcol];
        pd0 = d.x; pd1 = d.y;
    }

    for (int ks = 0; ks < IN_F / 32; ++ks) {
        __syncthreads();   // previous slab's LDS reads complete
        // split fp32 -> (hi,lo) bf16 pairs, packed dword LDS stores
        #pragma unroll
        for (int p = 0; p < 4; ++p) {
            #pragma unroll
            for (int j = 0; j < 4; j += 2) {
                const int idx = (lrow + 32 * p) * S + lcol + j;
                store_split(&sAh[idx], &sAl[idx], pa[p * 4 + j], pa[p * 4 + j + 1]);
                store_split(&sBh[idx], &sBl[idx], pb[p * 4 + j], pb[p * 4 + j + 1]);
            }
        }
        store_split(&sDh[drow * S + dcol], &sDl[drow * S + dcol], pd0, pd1);
        __syncthreads();

        // prefetch next slab (latency hidden by the WMMA block below)
        if (ks + 1 < IN_F / 32) {
            const int k = (ks + 1) * 32;
            #pragma unroll
            for (int p = 0; p < 4; ++p) {
                *(float4*)&pa[p * 4] = *(const float4*)&xA[(size_t)(lrow + 32 * p) * IN_F + k + lcol];
                *(float4*)&pb[p * 4] = *(const float4*)&wB[(size_t)(lrow + 32 * p) * IN_F + k + lcol];
            }
            if (drow < RANK) {
                float2 d = *(const float2*)&wD[(size_t)drow * IN_F + k + dcol];
                pd0 = d.x; pd1 = d.y;
            }
        }

        // WMMA block: D = A_hi*B_hi + A_hi*B_lo + A_lo*B_hi  (split-fp32 emulation)
        #pragma unroll
        for (int mt = 0; mt < 4; ++mt) {
            Frag ah, al;
            const int ao = (wr * 64 + mt * 16 + l16) * S + lh * 16; // A frag: contiguous 32B K-run
            ah.q[0] = *(const uint4*)&sAh[ao];
            ah.q[1] = *(const uint4*)&sAh[ao + 8];
            al.q[0] = *(const uint4*)&sAl[ao];
            al.q[1] = *(const uint4*)&sAl[ao + 8];
            #pragma unroll
            for (int nt = 0; nt < 2; ++nt) {
                Frag bh, bl;
                const int bo = (wc * 32 + nt * 16 + l16) * S + lh * 16; // B frag: contiguous 32B K-run
                bh.q[0] = *(const uint4*)&sBh[bo];
                bh.q[1] = *(const uint4*)&sBh[bo + 8];
                bl.q[0] = *(const uint4*)&sBl[bo];
                bl.q[1] = *(const uint4*)&sBl[bo + 8];
                v8f c = acc[mt][nt];
                c = __builtin_amdgcn_wmma_f32_16x16x32_bf16(false, ah.v, false, bh.v, (short)0, c, false, false);
                c = __builtin_amdgcn_wmma_f32_16x16x32_bf16(false, ah.v, false, bl.v, (short)0, c, false, false);
                c = __builtin_amdgcn_wmma_f32_16x16x32_bf16(false, al.v, false, bh.v, (short)0, c, false, false);
                acc[mt][nt] = c;
            }
            if (mt == wc) {   // fused LoRA down-projection: each wave owns one 16-row down tile
                Frag dh, dl;
                const int dco = l16 * S + lh * 16;
                dh.q[0] = *(const uint4*)&sDh[dco];
                dh.q[1] = *(const uint4*)&sDh[dco + 8];
                dl.q[0] = *(const uint4*)&sDl[dco];
                dl.q[1] = *(const uint4*)&sDl[dco + 8];
                dacc = __builtin_amdgcn_wmma_f32_16x16x32_bf16(false, ah.v, false, dh.v, (short)0, dacc, false, false);
                dacc = __builtin_amdgcn_wmma_f32_16x16x32_bf16(false, ah.v, false, dl.v, (short)0, dacc, false, false);
                dacc = __builtin_amdgcn_wmma_f32_16x16x32_bf16(false, al.v, false, dh.v, (short)0, dacc, false, false);
            }
        }
    }

    // share down[128][4] across waves (C/D layout: vgpr r -> row r+8*lh, col = l16)
    if (l16 < RANK) {
        #pragma unroll
        for (int r = 0; r < 8; ++r) {
            const int m = wr * 64 + wc * 16 + r + 8 * lh;
            sDown[m * RANK + l16] = dacc[r];
        }
    }
    __syncthreads();

    // epilogue: out = base + active * SCALE * (down @ Wu[idx]^T)
    float* op = out + ((size_t)b * TOKENS + row0) * OUT_F + col0;
    #pragma unroll
    for (int mt = 0; mt < 4; ++mt) {
        #pragma unroll
        for (int nt = 0; nt < 2; ++nt) {
            const int ncol = wc * 32 + nt * 16 + l16;
            float4 wu = make_float4(0.f, 0.f, 0.f, 0.f);
            if (active)
                wu = *(const float4*)&Wu[((size_t)safe * OUT_F + (col0 + ncol)) * RANK];
            #pragma unroll
            for (int r = 0; r < 8; ++r) {
                const int mrow = wr * 64 + mt * 16 + r + 8 * lh;
                float v = acc[mt][nt][r];
                if (active) {
                    float4 dn = *(const float4*)&sDown[mrow * RANK];
                    v += SCALE * (dn.x * wu.x + dn.y * wu.y + dn.z * wu.z + dn.w * wu.w);
                }
                op[(size_t)mrow * OUT_F + ncol] = v;
            }
        }
    }
}

extern "C" void kernel_launch(void* const* d_in, const int* in_sizes, int n_in,
                              void* d_out, int out_size, void* d_ws, size_t ws_size,
                              hipStream_t stream) {
    const float* x       = (const float*)d_in[0];
    const int*   lora_id = (const int*)d_in[1];
    const float* W       = (const float*)d_in[2];
    const float* Wd      = (const float*)d_in[3];
    const float* Wu      = (const float*)d_in[4];
    float*       out     = (float*)d_out;
    const int G = in_sizes[1];                 // number of chunks (16)
    dim3 grid(OUT_F / 128, TOKENS / 128, G);
    dim3 block(256);
    hipLaunchKernelGGL(lora_gemm_kernel, grid, block, 0, stream,
                       x, lora_id, W, Wd, Wu, out);
}
